// Task1MLP_15539191677329
// MI455X (gfx1250) — matrix operand order repaired and checked by reading, not compiled, not run on hardware
//
#include <hip/hip_runtime.h>
#include <hip/hip_bf16.h>

// ---------------------------------------------------------------------------
// MI455X (gfx1250) implementation.
//  - GEMMs via v_wmma_f32_16x16x32_bf16 (wave32, 16x16 tiles, f32 accum)
//  - A/W converted f32->bf16 during LDS staging (single pass over x)
//  - segment-sum pooling done deterministically via binary search on the
//    sorted E_map arrays (no float atomics), z2 (51 MB) is L2-resident.
// ---------------------------------------------------------------------------

#define IN_DIM 256
#define HIDDEN 128

typedef __attribute__((ext_vector_type(16))) __bf16 v16bf;
typedef __attribute__((ext_vector_type(8)))  float  v8f;

union FragBF {            // one WMMA 16-bf16-per-lane operand = 32 bytes
  uint4 u[2];
  v16bf v;
};

// ---------------------------------------------------------------------------
// Fused Linear(+bias)+ReLU:  Out = relu(A @ W + b)
//   A: [nrows x KDIM]  (f32 or bf16, row-major)
//   W: [KDIM x 128]    (f32, row-major; converted to bf16 in LDS, K-major)
//   Out: [nrows x 128] (bf16 or f32)
// Block: 256 threads = 8 waves; block tile 128(M) x 128(N); wave strip 16x128.
// ---------------------------------------------------------------------------
template <int KDIM, bool A_IS_F32, bool OUT_BF16>
__global__ __launch_bounds__(256)
void gemm_relu_kernel(const void* __restrict__ Aptr,
                      const float* __restrict__ W,
                      const float* __restrict__ bias,
                      void* __restrict__ Outptr,
                      int nrows)
{
  __shared__ __bf16 sA[128 * 32];     // A tile, [row][k]      (8 KB)
  __shared__ __bf16 sWt[128 * 32];    // W chunk, [col][k]     (8 KB) -> B frags contiguous
  __shared__ float  sC[8][16 * 128];  // per-wave repack       (64 KB)

  const int t    = threadIdx.x;
  const int wave = t >> 5;
  const int lane = t & 31;
  const int hl   = lane >> 4;   // which 16-lane half
  const int l16  = lane & 15;
  const int mbase = blockIdx.x * 128;

  const float*  Af = (const float*)Aptr;
  const __bf16* Ab = (const __bf16*)Aptr;

  v8f acc[8];
  #pragma unroll
  for (int n = 0; n < 8; ++n)
    #pragma unroll
    for (int r = 0; r < 8; ++r) acc[n][r] = 0.0f;

  for (int ks = 0; ks < KDIM / 32; ++ks) {
    const int K0 = ks * 32;
    __syncthreads();   // previous iteration's fragment reads complete

    // ---- stage A tile: 128 rows x 32 k (coalesced: 8 lanes cover one row) ----
    #pragma unroll
    for (int it = 0; it < 4; ++it) {
      const int idx = it * 256 + t;          // 0..1023
      const int row = idx >> 3;              // 0..127
      const int kq  = (idx & 7) * 4;         // 0,4,...,28
      size_t arow = (size_t)(mbase + row);
      if (arow >= (size_t)nrows) arow = (size_t)nrows - 1;   // clamp tail reads
      __bf16* dst = &sA[row * 32 + kq];
      if constexpr (A_IS_F32) {
        const float4 v = *(const float4*)(Af + arow * KDIM + K0 + kq);
        dst[0] = (__bf16)v.x; dst[1] = (__bf16)v.y;
        dst[2] = (__bf16)v.z; dst[3] = (__bf16)v.w;
      } else {
        *(uint2*)dst = *(const uint2*)(Ab + arow * KDIM + K0 + kq);
      }
    }

    // ---- stage W chunk transposed: sWt[col][k], converted f32->bf16 ----
    #pragma unroll
    for (int it = 0; it < 4; ++it) {
      const int idx = it * 256 + t;
      const int kr  = idx >> 5;              // 0..31
      const int nq  = (idx & 31) * 4;        // 0,4,...,124
      const float4 v = *(const float4*)(W + (size_t)(K0 + kr) * HIDDEN + nq);
      sWt[(nq + 0) * 32 + kr] = (__bf16)v.x;
      sWt[(nq + 1) * 32 + kr] = (__bf16)v.y;
      sWt[(nq + 2) * 32 + kr] = (__bf16)v.z;
      sWt[(nq + 3) * 32 + kr] = (__bf16)v.w;
    }
    __syncthreads();

    // ---- A fragment (contiguous K order, matching B):
    //      elem i -> K = 16*hl + i
    //      => two contiguous 16-byte LDS loads per lane.
    FragBF fa;
    const uint4* sA128 = (const uint4*)sA;
    const int rowA = wave * 16 + l16;
    fa.u[0] = sA128[(rowA * 32 + hl * 16) >> 3];
    fa.u[1] = sA128[(rowA * 32 + hl * 16 + 8) >> 3];

    // ---- 8 B fragments + WMMA (B 32x16: elem i -> K = 16*hl + i, contiguous) ----
    const uint4* sW128 = (const uint4*)sWt;
    #pragma unroll
    for (int n = 0; n < 8; ++n) {
      FragBF fb;
      const int col = n * 16 + l16;
      fb.u[0] = sW128[(col * 32 + hl * 16) >> 3];
      fb.u[1] = sW128[(col * 32 + hl * 16 + 8) >> 3];
      acc[n] = __builtin_amdgcn_wmma_f32_16x16x32_bf16(
                   false, fa.v, false, fb.v, (short)0, acc[n], false, false);
    }
  }

  // ---- epilogue: bias + ReLU, repack through per-wave LDS (C/D layout:
  //      elem r -> M = r + 8*hl, N = l16), then coalesced 16-byte stores.
  #pragma unroll
  for (int n = 0; n < 8; ++n) {
    const int col = n * 16 + l16;
    const float b = bias[col];
    #pragma unroll
    for (int r = 0; r < 8; ++r) {
      float v = acc[n][r] + b;
      v = v > 0.0f ? v : 0.0f;
      sC[wave][(r + hl * 8) * 128 + col] = v;
    }
  }
  // sC region is wave-private: same-wave DS ops are in-order (DScnt tracked).

  float*  Of = (float*)Outptr;
  __bf16* Ob = (__bf16*)Outptr;
  #pragma unroll
  for (int i = 0; i < 16; ++i) {
    const int c    = i * 32 + lane;   // float4 chunk id within 16x128 strip
    const int row  = c >> 5;
    const int col4 = (c & 31) * 4;
    const size_t m = (size_t)mbase + wave * 16 + row;
    if (m < (size_t)nrows) {
      const float4 v = *(const float4*)&sC[wave][row * 128 + col4];
      if constexpr (OUT_BF16) {
        union { __bf16 h[4]; uint2 u; } cv;
        cv.h[0] = (__bf16)v.x; cv.h[1] = (__bf16)v.y;
        cv.h[2] = (__bf16)v.z; cv.h[3] = (__bf16)v.w;
        *(uint2*)(Ob + m * HIDDEN + col4) = cv.u;
      } else {
        *(float4*)(Of + m * HIDDEN + col4) = v;
      }
    }
  }
}

// ---------------------------------------------------------------------------
// Fused gather + segment-sum + hadamard + Linear(128->1) + sigmoid.
// One wave per edge; E_map arrays are sorted, so each wave binary-searches its
// entry range (deterministic, no float atomics). z2 is L2-resident (51 MB).
// ---------------------------------------------------------------------------
__device__ __forceinline__ int lbound(const int* __restrict__ a, int n, int key) {
  int lo = 0, hi = n;
  while (lo < hi) {
    const int mid = (lo + hi) >> 1;
    if (a[mid] < key) lo = mid + 1; else hi = mid;
  }
  return lo;
}

__global__ __launch_bounds__(256)
void edge_pool_kernel(const float* __restrict__ z,
                      const int* __restrict__ V1, const int* __restrict__ V2,
                      const int* __restrict__ E1, const int* __restrict__ E2,
                      const float* __restrict__ Wl, const float* __restrict__ bl,
                      float* __restrict__ out, int n_edges, int m_entries)
{
  const int lane = threadIdx.x & 31;
  const int e = (blockIdx.x * blockDim.x + threadIdx.x) >> 5;
  if (e >= n_edges) return;            // wave-uniform branch

  const int d0 = lane * 4;             // 4 feature dims per lane (128/32)
  float4 a1 = make_float4(0.f, 0.f, 0.f, 0.f);
  float4 a2 = make_float4(0.f, 0.f, 0.f, 0.f);

  const int s1 = lbound(E1, m_entries, e);
  const int t1 = lbound(E1, m_entries, e + 1);
  for (int i = s1; i < t1; ++i) {
    const float4 v = *(const float4*)(z + (size_t)V1[i] * HIDDEN + d0);
    a1.x += v.x; a1.y += v.y; a1.z += v.z; a1.w += v.w;
  }
  const int s2 = lbound(E2, m_entries, e);
  const int t2 = lbound(E2, m_entries, e + 1);
  for (int i = s2; i < t2; ++i) {
    const float4 v = *(const float4*)(z + (size_t)V2[i] * HIDDEN + d0);
    a2.x += v.x; a2.y += v.y; a2.z += v.z; a2.w += v.w;
  }

  const float4 w = *(const float4*)(Wl + d0);
  float p = a1.x * a2.x * w.x + a1.y * a2.y * w.y
          + a1.z * a2.z * w.z + a1.w * a2.w * w.w;

  #pragma unroll
  for (int off = 16; off > 0; off >>= 1)
    p += __shfl_xor(p, off, 32);       // wave32 reduction

  if (lane == 0)
    out[e] = 1.0f / (1.0f + __expf(-(p + bl[0])));
}

// ---------------------------------------------------------------------------
extern "C" void kernel_launch(void* const* d_in, const int* in_sizes, int n_in,
                              void* d_out, int out_size, void* d_ws, size_t ws_size,
                              hipStream_t stream) {
  const float* x  = (const float*)d_in[0];
  const int*   V1 = (const int*)d_in[1];
  const int*   V2 = (const int*)d_in[2];
  const int*   E1 = (const int*)d_in[3];
  const int*   E2 = (const int*)d_in[4];
  const float* W1 = (const float*)d_in[5];
  const float* b1 = (const float*)d_in[6];
  const float* W2 = (const float*)d_in[7];
  const float* b2 = (const float*)d_in[8];
  const float* Wl = (const float*)d_in[9];
  const float* bl = (const float*)d_in[10];

  const int n_nodes   = in_sizes[0] / IN_DIM;
  const int m_entries = in_sizes[1];
  const int n_edges   = out_size;

  // workspace: z1 (bf16, 25.6 MB) | z2 (f32, 51.2 MB)
  size_t z1_bytes = ((size_t)n_nodes * HIDDEN * sizeof(__bf16) + 255) & ~(size_t)255;
  __bf16* z1b = (__bf16*)d_ws;
  float*  z2  = (float*)((char*)d_ws + z1_bytes);

  const int mtiles = (n_nodes + 127) / 128;
  gemm_relu_kernel<IN_DIM, true, true>
      <<<mtiles, 256, 0, stream>>>(x, W1, b1, z1b, n_nodes);
  gemm_relu_kernel<HIDDEN, false, false>
      <<<mtiles, 256, 0, stream>>>(z1b, W2, b2, z2, n_nodes);

  const int eblocks = (n_edges + 7) / 8;   // 8 waves (edges) per 256-thr block
  edge_pool_kernel<<<eblocks, 256, 0, stream>>>(z2, V1, V2, E1, E2, Wl, bl,
                                                (float*)d_out, n_edges, m_entries);
}
